// GatedMultiplicativeSelfAttention_2723009266486
// MI455X (gfx1250) — matrix operand order repaired and checked by reading, not compiled, not run on hardware
//
#include <hip/hip_runtime.h>

// Problem constants
#define Bb 8
#define Ss 2048
#define Dd 512
#define Hh 512

typedef __attribute__((ext_vector_type(16))) __bf16 v16bf;
typedef __attribute__((ext_vector_type(8)))  float  v8f;
typedef int v4i_t __attribute__((vector_size(16)));   // builtin's pointee type per diagnostic

struct FragBF { union { v16bf v; uint4 q[2]; }; };

// ---- CDNA5 async-copy availability (guarded; fallback keeps compile clean) ----
#if defined(__has_builtin)
#  if __has_builtin(__builtin_amdgcn_global_load_async_to_lds_b128)
#    define USE_ASYNC 1
#  endif
#endif
#ifndef USE_ASYNC
#  define USE_ASYNC 0
#endif

__device__ inline void cp_b128(const unsigned short* g, unsigned short* l){
#if USE_ASYNC
  __builtin_amdgcn_global_load_async_to_lds_b128((v4i_t*)g, (v4i_t*)l, 0, 0);
#else
  *(uint4*)l = *(const uint4*)g;
#endif
}
template<int N> __device__ inline void wait_async(){
#if USE_ASYNC
#  if __has_builtin(__builtin_amdgcn_s_wait_asynccnt)
  __builtin_amdgcn_s_wait_asynccnt((unsigned short)N);
#  else
  asm volatile("s_wait_asynccnt %0" :: "i"(N));
#  endif
#endif
}

__device__ inline unsigned short f2bf(float f){
  unsigned int u = __float_as_uint(f);
  u += 0x7FFFu + ((u >> 16) & 1u);           // round-to-nearest-even
  return (unsigned short)(u >> 16);
}
__device__ inline float bf2f(unsigned short h){
  return __uint_as_float(((unsigned int)h) << 16);
}
__device__ inline float sigm(float x){ return 1.f / (1.f + __expf(-x)); }
__device__ inline v8f vzero(){ v8f z = {0.f,0.f,0.f,0.f,0.f,0.f,0.f,0.f}; return z; }

// A-matrix 16x32 bf16 fragment (row-major source, stride lda elements).
// Lanes 0-15: row M=lane, K = k0+0..7 and k0+16..23; lanes 16-31: K = k0+8..15 and k0+24..31.
__device__ inline void loadA(FragBF& f, const unsigned short* A, int lda, int row0, int k0, int lane){
  const int m  = lane & 15;
  const int kh = (lane >> 4) << 3;
  const unsigned short* p = A + (size_t)(row0 + m) * lda + k0 + kh;
  f.q[0] = *reinterpret_cast<const uint4*>(p);
  f.q[1] = *reinterpret_cast<const uint4*>(p + 16);
}
// B-matrix 32x16 bf16 fragment from weight stored [N,K] row-major (stride ldw).
// Lane n=lane&15 owns column n; lanes 0-15 hold K=k0+0..7 and k0+16..23, lanes 16-31 hold K=k0+8..15 and k0+24..31.
__device__ inline void loadB(FragBF& f, const unsigned short* W, int ldw, int n0, int k0, int lane){
  const int n  = lane & 15;
  const int kh = (lane >> 4) << 3;
  const unsigned short* p = W + (size_t)(n0 + n) * ldw + k0 + kh;
  f.q[0] = *reinterpret_cast<const uint4*>(p);
  f.q[1] = *reinterpret_cast<const uint4*>(p + 16);
}
__device__ inline v8f wmma_bf16(const FragBF& a, const FragBF& b, v8f c){
  return __builtin_amdgcn_wmma_f32_16x16x32_bf16(false, a.v, false, b.v, (short)0, c, false, false);
}

// ---------------- conversion kernels ----------------
__global__ void cvt_f32_bf16(const float* __restrict__ src, unsigned short* __restrict__ dst, int n){
  int i = blockIdx.x * blockDim.x + threadIdx.x;
  if (i < n) dst[i] = f2bf(src[i]);
}

// x -> rinb[:, :512] (stride 1024) and transposed xbT [B, D, S]
__global__ void prep_x(const float* __restrict__ x, unsigned short* __restrict__ rinb,
                       unsigned short* __restrict__ xbT){
  int i = blockIdx.x * blockDim.x + threadIdx.x;
  if (i >= Bb * Ss * Dd) return;
  int d = i & (Dd - 1);
  int s = (i >> 9) & (Ss - 1);
  int b = i >> 20;
  unsigned short v = f2bf(x[i]);
  rinb[((size_t)(b * Ss + s) << 10) + d] = v;
  xbT [((size_t)(b * Dd + d) << 11) + s] = v;
}

// ---------------- generic WMMA GEMM: C = epilogue(A @ W^T) ----------------
// Block = 128 threads = 4 waves; block tile 64(M) x 64(N); K in steps of 32.
// B tile (shared by all 4 waves) is staged in LDS via double-buffered async
// copies (ASYNCcnt); A fragments stream from global (each used once per block).
// MODE 0: store bf16          MODE 1: store bf16( aux * sigmoid(acc) )
// MODE 2: store bf16(acc + bias[n])   MODE 3: store f32(acc + bias[n] + aux_f32)
template<int MODE>
__global__ __launch_bounds__(128) void gemm_bf16(
    const unsigned short* __restrict__ A, int lda, long long sA,
    const unsigned short* __restrict__ Wt, int ldw, long long sW,
    void* __restrict__ Out, int ldo, long long sO,
    const float* __restrict__ bias,
    const void* __restrict__ Aux, int ldaux, long long sAux,
    int Kdim)
{
  __shared__ unsigned short bstage[2][64 * 32];   // 2 x 4KB B tiles
  const int tid  = threadIdx.x;
  const int lane = tid & 31, wave = tid >> 5;
  const int half = lane >> 4, ln = lane & 15;
  const int zz = blockIdx.z;
  A  += (size_t)zz * sA;
  Wt += (size_t)zz * sW;
  const int m0 = blockIdx.x * 64 + wave * 16;
  const int n0 = blockIdx.y * 64;

  // per-thread B-tile copy assignment: 128 threads x 32B = 4KB tile
  const int bn = tid >> 1;            // 0..63 (row n within tile)
  const int bk = (tid & 1) << 4;      // 0 or 16 (k within tile)
  const unsigned short* bsrc = Wt + (size_t)(n0 + bn) * ldw + bk;
  unsigned short* bdst[2] = { &bstage[0][bn * 32 + bk], &bstage[1][bn * 32 + bk] };

  v8f acc[4];
#pragma unroll
  for (int nt = 0; nt < 4; ++nt) acc[nt] = vzero();

  const int nk = Kdim >> 5;
  cp_b128(bsrc,     bdst[0]);
  cp_b128(bsrc + 8, bdst[0] + 8);
  for (int kk = 0; kk < nk; ++kk){
    const int k0 = kk << 5;
    if (kk + 1 < nk){                          // prefetch next B tile into other buffer
      const unsigned short* s = bsrc + (size_t)(k0 + 32);
      unsigned short* d = bdst[(kk + 1) & 1];
      cp_b128(s,     d);
      cp_b128(s + 8, d + 8);
      wait_async<2>();                         // current buffer (first 2 ops) complete
    } else {
      wait_async<0>();
    }
    __syncthreads();
    __builtin_prefetch(A + (size_t)(m0 + ln) * lda + k0 + 64, 0, 1);  // A stream ahead
    FragBF a; loadA(a, A, lda, m0, k0, lane);
    const unsigned short* bb = bstage[kk & 1];
#pragma unroll
    for (int nt = 0; nt < 4; ++nt){
      FragBF b; loadB(b, bb, 32, nt * 16, 0, lane);   // ds_load from staged tile
      acc[nt] = wmma_bf16(a, b, acc[nt]);
    }
    __syncthreads();
  }

#pragma unroll
  for (int nt = 0; nt < 4; ++nt){
#pragma unroll
    for (int i = 0; i < 8; ++i){
      const int m = m0 + i + 8 * half;          // C/D layout: VGPR i, lane-half selects M
      const int n = n0 + nt * 16 + ln;          // column striped across 16 lanes
      const float v = acc[nt][i];
      const size_t oi = (size_t)m * ldo + n;
      if constexpr (MODE == 0){
        ((unsigned short*)Out + (size_t)zz * sO)[oi] = f2bf(v);
      } else if constexpr (MODE == 1){
        const unsigned short* ax = (const unsigned short*)Aux + (size_t)zz * sAux;
        float r = bf2f(ax[(size_t)m * ldaux + n]);
        ((unsigned short*)Out + (size_t)zz * sO)[oi] = f2bf(r * sigm(v));
      } else if constexpr (MODE == 2){
        ((unsigned short*)Out + (size_t)zz * sO)[oi] = f2bf(v + bias[n]);
      } else {
        const float* ax = (const float*)Aux + (size_t)zz * sAux;
        ((float*)Out + (size_t)zz * sO)[oi] = v + bias[n] + ax[(size_t)m * ldaux + n];
      }
    }
  }
}

// ---------------- attention ----------------
// One 16x64 score tile per wave per j-block: S = Q(16x512) @ K_j(512x64)^T
__device__ inline void score_tile(v8f acc[4], const unsigned short* Q, const unsigned short* Kz,
                                  int q0, int jb, int lane){
#pragma unroll
  for (int nt = 0; nt < 4; ++nt) acc[nt] = vzero();
  for (int k0 = 0; k0 < Dd; k0 += 32){
    FragBF a; loadA(a, Q, Dd, q0, k0, lane);
#pragma unroll
    for (int nt = 0; nt < 4; ++nt){
      FragBF b; loadB(b, Kz, 2 * Dd, jb + nt * 16, k0, lane);
      acc[nt] = wmma_bf16(a, b, acc[nt]);
    }
  }
}

// Pass 1: streaming row max + sumexp. grid (S/64, B), 128 threads (4 waves x 16 rows).
__global__ __launch_bounds__(128) void attn_pass1(const unsigned short* __restrict__ wxq,
                                                  const unsigned short* __restrict__ keys,
                                                  float* __restrict__ rowm, float* __restrict__ rowl){
  const int lane = threadIdx.x & 31, wave = threadIdx.x >> 5;
  const int half = lane >> 4, ln = lane & 15;
  const int b = blockIdx.y;
  const unsigned short* Q  = wxq  + (size_t)b * Ss * Dd;
  const unsigned short* Kz = keys + (size_t)b * Ss * (2 * Dd);
  const int q0 = blockIdx.x * 64 + wave * 16;

  float mrow[8], lrow[8];
#pragma unroll
  for (int v = 0; v < 8; ++v){ mrow[v] = -1e30f; lrow[v] = 0.f; }

  for (int jb = 0; jb < Ss; jb += 64){
    v8f acc[4];
    score_tile(acc, Q, Kz, q0, jb, lane);
#pragma unroll
    for (int nt = 0; nt < 4; ++nt){
#pragma unroll
      for (int v = 0; v < 8; ++v){
        int gi = q0 + v + 8 * half;
        int gj = jb + nt * 16 + ln;
        if (gi == gj) acc[nt][v] = -1e30f;      // mask diagonal
      }
    }
#pragma unroll
    for (int v = 0; v < 8; ++v){
      float bm = fmaxf(fmaxf(acc[0][v], acc[1][v]), fmaxf(acc[2][v], acc[3][v]));
#pragma unroll
      for (int o = 8; o >= 1; o >>= 1) bm = fmaxf(bm, __shfl_xor(bm, o, 32));
      float mn = fmaxf(mrow[v], bm);
      float s = __expf(acc[0][v] - mn) + __expf(acc[1][v] - mn)
              + __expf(acc[2][v] - mn) + __expf(acc[3][v] - mn);
#pragma unroll
      for (int o = 8; o >= 1; o >>= 1) s += __shfl_xor(s, o, 32);
      lrow[v] = lrow[v] * __expf(mrow[v] - mn) + s;
      mrow[v] = mn;
    }
  }
  if (ln == 0){
#pragma unroll
    for (int v = 0; v < 8; ++v){
      int gi = q0 + v + 8 * half;
      rowm[(size_t)b * Ss + gi] = mrow[v];
      rowl[(size_t)b * Ss + gi] = lrow[v];
    }
  }
}

// Pass 2: write normalized probabilities P (bf16, [B,S,S] -> L2-resident 67MB).
__global__ __launch_bounds__(128) void attn_pass2(const unsigned short* __restrict__ wxq,
                                                  const unsigned short* __restrict__ keys,
                                                  const float* __restrict__ rowm,
                                                  const float* __restrict__ rowl,
                                                  unsigned short* __restrict__ P){
  const int lane = threadIdx.x & 31, wave = threadIdx.x >> 5;
  const int half = lane >> 4, ln = lane & 15;
  const int b = blockIdx.y;
  const unsigned short* Q  = wxq  + (size_t)b * Ss * Dd;
  const unsigned short* Kz = keys + (size_t)b * Ss * (2 * Dd);
  const int q0 = blockIdx.x * 64 + wave * 16;

  float mrow[8], rinv[8];
#pragma unroll
  for (int v = 0; v < 8; ++v){
    int gi = q0 + v + 8 * half;
    mrow[v] = rowm[(size_t)b * Ss + gi];
    rinv[v] = 1.f / rowl[(size_t)b * Ss + gi];
  }
  for (int jb = 0; jb < Ss; jb += 64){
    v8f acc[4];
    score_tile(acc, Q, Kz, q0, jb, lane);
#pragma unroll
    for (int nt = 0; nt < 4; ++nt){
#pragma unroll
      for (int v = 0; v < 8; ++v){
        int gi = q0 + v + 8 * half;
        int gj = jb + nt * 16 + ln;
        float p = (gi == gj) ? 0.f : __expf(acc[nt][v] - mrow[v]) * rinv[v];
        P[((size_t)b * Ss + gi) * Ss + gj] = f2bf(p);
      }
    }
  }
}

// ---------------- GRU recurrence ----------------
// grid=2 (dir), 1024 threads = 32 waves. h [16x512] bf16 in LDS (rows 8..15 = zero pad).
// Per step: gh = h @ Whh^T (16x1536) via WMMA, 3 n-tiles per wave; then elementwise gates.
__global__ __launch_bounds__(1024) void gru_kernel(
    const unsigned short* __restrict__ xwf, const unsigned short* __restrict__ xwbk,
    const unsigned short* __restrict__ whhf, const unsigned short* __restrict__ whhb,
    const float* __restrict__ bhhf, const float* __restrict__ bhhb,
    unsigned short* __restrict__ hcat)
{
  __shared__ unsigned short hbuf[16 * 512];   // 16 KB
  __shared__ float ghbuf[8 * 1536];           // 48 KB
  const int dir = blockIdx.x;
  const unsigned short* xw  = dir ? xwbk : xwf;
  const unsigned short* Whh = dir ? whhb : whhf;
  const float*          bhh = dir ? bhhb : bhhf;
  const int colOff = dir * Hh;
  const int tid = threadIdx.x, lane = tid & 31, wave = tid >> 5;
  const int half = lane >> 4, ln = lane & 15;

  for (int i = tid; i < 16 * 512; i += 1024) hbuf[i] = 0;
  __syncthreads();

  for (int t = 0; t < Ss; ++t){
    const int tt = dir ? (Ss - 1 - t) : t;
    // gh = h @ Whh^T  (M=16 padded, N=1536, K=512)
    v8f acc[3];
#pragma unroll
    for (int q = 0; q < 3; ++q) acc[q] = vzero();
    for (int k0 = 0; k0 < Hh; k0 += 32){
      FragBF a; loadA(a, (const unsigned short*)hbuf, 512, 0, k0, lane);
#pragma unroll
      for (int q = 0; q < 3; ++q){
        FragBF b; loadB(b, Whh, Hh, (wave * 3 + q) * 16, k0, lane);
        acc[q] = wmma_bf16(a, b, acc[q]);
      }
    }
    if (half == 0){                       // rows 0..7 are the real batch
#pragma unroll
      for (int q = 0; q < 3; ++q){
#pragma unroll
        for (int i = 0; i < 8; ++i){
          int n = (wave * 3 + q) * 16 + ln;
          ghbuf[i * 1536 + n] = acc[q][i] + bhh[n];
        }
      }
    }
    __syncthreads();
    // gates: r,z,n + state update
    for (int e = tid; e < Bb * Hh; e += 1024){
      int b = e >> 9, j = e & 511;
      size_t xi = ((size_t)b * Ss + tt) * 1536;
      float gr = ghbuf[b * 1536 + j];
      float gz = ghbuf[b * 1536 + 512 + j];
      float gn = ghbuf[b * 1536 + 1024 + j];
      float r = sigm(bf2f(xw[xi + j]) + gr);
      float z = sigm(bf2f(xw[xi + 512 + j]) + gz);
      float n = tanhf(bf2f(xw[xi + 1024 + j]) + r * gn);
      float hp = bf2f(hbuf[b * 512 + j]);
      float hn = (1.f - z) * n + z * hp;
      unsigned short hv = f2bf(hn);
      hbuf[b * 512 + j] = hv;
      hcat[((size_t)b * Ss + tt) * 1024 + colOff + j] = hv;
    }
    __syncthreads();
  }
}

// ---------------- workspace layout (bytes) ----------------
static constexpr size_t OFF_RINB = 0;                        // 33,554,432  [B*S,1024] bf16
static constexpr size_t OFF_WXB  = 33554432ull;              // 16,777,216  [B*S,512]  bf16
static constexpr size_t OFF_RGB  = 50331648ull;              // 33,554,432  gated rin (hcat aliases)
static constexpr size_t OFF_P    = 83886080ull;              // 67,108,864  probs (xw_b aliases)
static constexpr size_t OFF_XBT  = 150994944ull;             // 16,777,216  x^T bf16
static constexpr size_t OFF_WTS  = 167772160ull;             // 13,107,200  bf16 weights
static constexpr size_t OFF_ROWM = 180879360ull;             // 65,536
static constexpr size_t OFF_ROWL = 180944896ull;             // 65,536
// xw_f aliases [RINB..RINB+50,331,648) ; xw_b aliases P ; hcat aliases RGB

extern "C" void kernel_launch(void* const* d_in, const int* in_sizes, int n_in,
                              void* d_out, int out_size, void* d_ws, size_t ws_size,
                              hipStream_t stream)
{
  (void)in_sizes; (void)n_in; (void)out_size; (void)ws_size;
  const float* x    = (const float*)d_in[0];
  const float* W    = (const float*)d_in[1];
  const float* Wg   = (const float*)d_in[2];
  const float* Wihf = (const float*)d_in[3];
  const float* Whhf = (const float*)d_in[4];
  const float* bihf = (const float*)d_in[5];
  const float* bhhf = (const float*)d_in[6];
  const float* Wihb = (const float*)d_in[7];
  const float* Whhb = (const float*)d_in[8];
  const float* bihb = (const float*)d_in[9];
  const float* bhhb = (const float*)d_in[10];
  const float* Wp   = (const float*)d_in[11];
  const float* bp   = (const float*)d_in[12];

  char* ws = (char*)d_ws;
  unsigned short* rinb = (unsigned short*)(ws + OFF_RINB);
  unsigned short* wxb  = (unsigned short*)(ws + OFF_WXB);
  unsigned short* rgb  = (unsigned short*)(ws + OFF_RGB);
  unsigned short* Pp   = (unsigned short*)(ws + OFF_P);
  unsigned short* xbT  = (unsigned short*)(ws + OFF_XBT);
  unsigned short* wts  = (unsigned short*)(ws + OFF_WTS);
  float* rowm = (float*)(ws + OFF_ROWM);
  float* rowl = (float*)(ws + OFF_ROWL);
  unsigned short* Wb    = wts;
  unsigned short* Wgb   = wts + 262144;
  unsigned short* Wihfb = wts + 1310720;
  unsigned short* Whhfb = wts + 2883584;
  unsigned short* Wihbb = wts + 3670016;
  unsigned short* Whhbb = wts + 5242880;
  unsigned short* Wpb   = wts + 6029312;
  unsigned short* xwf  = rinb;   // alias: rinb+wxb dead after gate GEMM
  unsigned short* xwb  = Pp;     // alias: P dead after P@V
  unsigned short* hcat = rgb;    // alias: rgb dead after xw GEMMs

  auto cv = [&](const float* s, unsigned short* d, int n){
    cvt_f32_bf16<<<(n + 255) / 256, 256, 0, stream>>>(s, d, n);
  };
  cv(W,    Wb,    512 * 512);
  cv(Wg,   Wgb,   1024 * 1024);
  cv(Wihf, Wihfb, 1536 * 1024);
  cv(Whhf, Whhfb, 1536 * 512);
  cv(Wihb, Wihbb, 1536 * 1024);
  cv(Whhb, Whhbb, 1536 * 512);
  cv(Wp,   Wpb,   512 * 1024);
  prep_x<<<(Bb * Ss * Dd + 255) / 256, 256, 0, stream>>>(x, rinb, xbT);

  // Wx = x @ W^T : M=16384, N=512, K=512
  gemm_bf16<0><<<dim3(256, 8, 1), 128, 0, stream>>>(rinb, 1024, 0, Wb, 512, 0,
      wxb, 512, 0, nullptr, nullptr, 0, 0, 512);
  // attention softmax statistics + probabilities
  attn_pass1<<<dim3(32, 8), 128, 0, stream>>>(wxb, rinb, rowm, rowl);
  attn_pass2<<<dim3(32, 8), 128, 0, stream>>>(wxb, rinb, rowm, rowl, Pp);
  // c = P @ x : batched, M=2048, N=512, K=2048 ; writes rinb cols 512..1023
  gemm_bf16<0><<<dim3(32, 8, 8), 128, 0, stream>>>(Pp, 2048, (long long)Ss * Ss,
      xbT, 2048, (long long)Dd * Ss,
      (void*)(rinb + 512), 1024, (long long)Ss * 1024, nullptr, nullptr, 0, 0, 2048);
  // gated concat: rgb = rinb * sigmoid(rinb @ Wg^T) : M=16384, N=1024, K=1024
  gemm_bf16<1><<<dim3(256, 16, 1), 128, 0, stream>>>(rinb, 1024, 0, Wgb, 1024, 0,
      rgb, 1024, 0, nullptr, rinb, 1024, 0, 1024);
  // GRU input projections: M=16384, N=1536, K=1024
  gemm_bf16<2><<<dim3(256, 24, 1), 128, 0, stream>>>(rgb, 1024, 0, Wihfb, 1024, 0,
      xwf, 1536, 0, bihf, nullptr, 0, 0, 1024);
  gemm_bf16<2><<<dim3(256, 24, 1), 128, 0, stream>>>(rgb, 1024, 0, Wihbb, 1024, 0,
      xwb, 1536, 0, bihb, nullptr, 0, 0, 1024);
  // bidirectional GRU recurrence (both directions in parallel)
  gru_kernel<<<2, 1024, 0, stream>>>(xwf, xwb, Whhfb, Whhbb, bhhf, bhhb, hcat);
  // out = x + hcat @ Wp^T + bp : M=16384, N=512, K=1024
  gemm_bf16<3><<<dim3(256, 8, 1), 128, 0, stream>>>(hcat, 1024, 0, Wpb, 1024, 0,
      d_out, 512, 0, bp, x, 512, 0, 1024);
}
